// BaseModel_87316685128012
// MI455X (gfx1250) — matrix operand order repaired and checked by reading, not compiled, not run on hardware
//
#include <hip/hip_runtime.h>

typedef __attribute__((ext_vector_type(16))) __bf16 v16bf;
typedef __attribute__((ext_vector_type(8)))  __bf16 bf16x8;
typedef __attribute__((ext_vector_type(8)))  float  v8f;

union ABu { v16bf v; bf16x8 h[2]; };

#define BATCH 256
#define SEEDL 120
#define PREDL 24
#define DIM   135
#define DP    160     // DIM padded to mult of 32
#define HDIM  1024
#define S3    3072
#define FBINS 61
#define INPK  1088    // 1069 padded
#define CHB   64      // attention batch chunk

#define WMMA_BF16(a, b, c) \
  __builtin_amdgcn_wmma_f32_16x16x32_bf16(false, (a), false, (b), (short)0, (c), false, false)

// A 16x32 bf16 frag: lanes 0-15 row m hold K {+0..7, +16..23}; lanes 16-31 hold +8.
__device__ __forceinline__ v16bf ldA(const __bf16* p){
  ABu u; u.h[0] = *(const bf16x8*)p; u.h[1] = *(const bf16x8*)(p + 16);
  return u.v;
}
// B 32x16 bf16 frag (W[N][K] row per lane-col): lanes 0-15 K {+0..7, +16..23}; lanes 16-31 hold +8.
__device__ __forceinline__ v16bf ldB(const __bf16* p){
  ABu u; u.h[0] = *(const bf16x8*)p; u.h[1] = *(const bf16x8*)(p + 16);
  return u.v;
}

// ---------------- generic bf16 WMMA GEMM: C = A*W^T + bias, epilogue ----------------
// A: MxK bf16 (lda), W: NxK bf16 (ldw), N multiple of 64, K multiple of 32.
// Each wave computes a 32x64 tile (2 A-frags x 4 B-frags, 8 WMMAs per 32-K step),
// software-pipelined with incremental per-lane pointers (no per-step address math).
// WN==4: waves 2(M)x4(N), block tile 64x256, grid (N/256, M/64).
// WN==1: waves 8(M)x1(N), block tile 256x64, grid (N/64, M/256).
// Cf optional f32 out, Cb optional bf16 out; cols >= nvalid masked.
// act: 0=none, 1=relu, 2=tanh.
template<int WN>
__global__ __launch_bounds__(256, 1)
void k_gemm(const __bf16* __restrict__ A, int lda,
            const __bf16* __restrict__ W, int ldw,
            const float* __restrict__ bias,
            float* __restrict__ Cf, int ldc,
            __bf16* __restrict__ Cb, int ldcb,
            int K, int nvalid, int act)
{
  int tid  = threadIdx.x;
  int lane = tid & 31;
  int wave = __builtin_amdgcn_readfirstlane(tid >> 5);   // scalar wave id
  int wm, wn;
  if (WN == 4){ wm = wave >> 2; wn = wave & 3; }
  else        { wm = wave;      wn = 0;        }
  const int mBlkWaves = (WN == 4) ? 2 : 8;
  int m0 = (blockIdx.y * mBlkWaves + wm) * 32;
  int n0 = (blockIdx.x * WN + wn) * 64;

  int l15 = lane & 15;
  int hi  = lane >> 4;

  // per-lane streaming pointers, advanced 32 elements (64B) per k-step
  const __bf16* pa0 = A + (size_t)(m0 + l15) * lda + (hi << 3);
  const __bf16* pa1 = pa0 + (size_t)16 * lda;
  const __bf16* pb0 = W + (size_t)(n0 + l15) * ldw + (hi << 3);
  const __bf16* pb1 = pb0 + (size_t)16 * ldw;
  const __bf16* pb2 = pb0 + (size_t)32 * ldw;
  const __bf16* pb3 = pb0 + (size_t)48 * ldw;

  v8f acc0 = {}, acc1 = {}, acc2 = {}, acc3 = {};
  v8f acc4 = {}, acc5 = {}, acc6 = {}, acc7 = {};

  v16bf a0 = ldA(pa0), a1 = ldA(pa1);
  v16bf b0 = ldB(pb0), b1 = ldB(pb1), b2 = ldB(pb2), b3 = ldB(pb3);
  pa0 += 32; pa1 += 32; pb0 += 32; pb1 += 32; pb2 += 32; pb3 += 32;

  for (int k0 = 32; k0 < K; k0 += 32) {
    v16bf na0 = ldA(pa0), na1 = ldA(pa1);
    v16bf nb0 = ldB(pb0), nb1 = ldB(pb1), nb2 = ldB(pb2), nb3 = ldB(pb3);
    pa0 += 32; pa1 += 32; pb0 += 32; pb1 += 32; pb2 += 32; pb3 += 32;
    acc0 = WMMA_BF16(a0, b0, acc0);
    acc1 = WMMA_BF16(a0, b1, acc1);
    acc2 = WMMA_BF16(a0, b2, acc2);
    acc3 = WMMA_BF16(a0, b3, acc3);
    acc4 = WMMA_BF16(a1, b0, acc4);
    acc5 = WMMA_BF16(a1, b1, acc5);
    acc6 = WMMA_BF16(a1, b2, acc6);
    acc7 = WMMA_BF16(a1, b3, acc7);
    a0 = na0; a1 = na1; b0 = nb0; b1 = nb1; b2 = nb2; b3 = nb3;
  }
  acc0 = WMMA_BF16(a0, b0, acc0);
  acc1 = WMMA_BF16(a0, b1, acc1);
  acc2 = WMMA_BF16(a0, b2, acc2);
  acc3 = WMMA_BF16(a0, b3, acc3);
  acc4 = WMMA_BF16(a1, b0, acc4);
  acc5 = WMMA_BF16(a1, b1, acc5);
  acc6 = WMMA_BF16(a1, b2, acc6);
  acc7 = WMMA_BF16(a1, b3, acc7);

  // C layout: lane<16 -> rows +0..7; lane>=16 -> rows +8..15 of each 16-row tile.
  int rhalf = hi << 3;
  v8f accs[8] = {acc0, acc1, acc2, acc3, acc4, acc5, acc6, acc7};
  #pragma unroll
  for (int mt = 0; mt < 2; mt++) {
    int rb = m0 + mt * 16 + rhalf;
    #pragma unroll
    for (int c = 0; c < 4; c++) {
      int col = n0 + c * 16 + l15;
      bool wr = col < nvalid;
      float bv = (wr && bias) ? bias[col] : 0.f;
      #pragma unroll
      for (int r = 0; r < 8; r++) {
        float v = accs[mt*4 + c][r] + bv;
        if (act == 1) v = fmaxf(v, 0.f);
        else if (act == 2) v = tanhf(v);
        if (wr) {
          if (Cf) Cf[(size_t)(rb + r) * ldc  + col] = v;
          if (Cb) Cb[(size_t)(rb + r) * ldcb + col] = (__bf16)v;
        }
      }
    }
  }
}

// ---------------- weight prep ----------------
__global__ void k_cvt_pad(__bf16* dst, const float* src, int K, int Kp){
  int n = blockIdx.x;
  for (int k = threadIdx.x; k < Kp; k += blockDim.x)
    dst[(size_t)n*Kp + k] = (__bf16)(k < K ? src[(size_t)n*K + k] : 0.f);
}
// spl_W1 [15][1069][64] -> [15][64][1088] (N-major, K padded)
__global__ void k_cvt_w1(__bf16* dst, const float* src){
  int j = blockIdx.x >> 6, n = blockIdx.x & 63;
  for (int k = threadIdx.x; k < INPK; k += blockDim.x){
    float v = (k < 1069) ? src[((size_t)j*1069 + k)*64 + n] : 0.f;
    dst[((size_t)j*64 + n)*INPK + k] = (__bf16)v;
  }
}
// spl_W2 [15][64][9] -> [15][64][64] (N padded to 64 with zero rows)
__global__ void k_cvt_w2(__bf16* dst, const float* src){
  int j = blockIdx.x >> 6, n = blockIdx.x & 63;
  int k = threadIdx.x; // 64
  float v = (n < 9) ? src[((size_t)j*64 + k)*9 + n] : 0.f;
  dst[((size_t)j*64 + n)*64 + k] = (__bf16)v;
}
__global__ void k_cvt_seed(__bf16* dst, const float* poses){
  int bt = blockIdx.x; int b = bt / SEEDL, t = bt % SEEDL;
  int d = threadIdx.x; // 160
  float v = (d < DIM) ? poses[((size_t)b*(SEEDL+PREDL) + t)*DIM + d] : 0.f;
  dst[(size_t)bt*DP + d] = (__bf16)v;
}
__global__ void k_zero_f32(float* p, int n){ int i = blockIdx.x*256 + threadIdx.x; if (i < n) p[i] = 0.f; }
__global__ void k_zero_b16(__bf16* p, int n){ int i = blockIdx.x*256 + threadIdx.x; if (i < n) p[i] = (__bf16)0.f; }
__global__ void k_init_x(float* xf, __bf16* xb, const float* poses){
  int b = blockIdx.x, d = threadIdx.x; // 160
  if (d < DIM){
    float v = poses[((size_t)b*(SEEDL+PREDL) + (SEEDL-1))*DIM + d];
    xf[(size_t)b*DIM + d] = v;
    xb[(size_t)b*DP  + d] = (__bf16)v;
  } else xb[(size_t)b*DP + d] = (__bf16)0.f;
}

// ---------------- rfft(seed).real as exact-phase cosine transform ----------------
__global__ void k_freq(const float* __restrict__ poses, __bf16* __restrict__ freqb){
  int bf_ = blockIdx.x; int b = bf_ / FBINS, f = bf_ % FBINS;
  __shared__ float ct[SEEDL];
  int t = threadIdx.x; // 160
  if (t < SEEDL){
    int m = (f * t) % SEEDL;  // exact integer phase
    ct[t] = cosf(6.2831853071795864769f * (float)m / (float)SEEDL);
  }
  __syncthreads();
  float s = 0.f;
  if (t < DIM){
    const float* src = poses + ((size_t)b*(SEEDL+PREDL))*DIM + t;
    for (int tt = 0; tt < SEEDL; tt++) s += src[(size_t)tt*DIM] * ct[tt];
  }
  if (t < DP) freqb[(size_t)bf_*DP + t] = (__bf16)((t < DIM) ? s : 0.f);
}

// ---------------- attention: pooled context per (batch, head) ----------------
// mean_s(softmax(QK^T/16) V) = (colsum(attn)/61) . V  -> obar
__global__ void k_attn(const __bf16* __restrict__ qkv, __bf16* __restrict__ obar, int bbase){
  int bl = blockIdx.x >> 2, h = blockIdx.x & 3;
  int gb = bbase + bl;
  const __bf16* base = qkv + (size_t)bl * FBINS * S3;
  __shared__ __bf16 ksh[FBINS * 256];
  __shared__ float  sc[FBINS * 64];
  __shared__ float  wsum[64];
  int tid = threadIdx.x; // 256
  for (int i = tid; i < FBINS*256; i += 256){
    int j = i >> 8, d = i & 255;
    ksh[i] = base[(size_t)j*S3 + HDIM + h*256 + d];
  }
  __syncthreads();
  for (int p = tid; p < FBINS*FBINS; p += 256){
    int i = p / FBINS, j = p % FBINS;
    const __bf16* q = base + (size_t)i*S3 + h*256;
    float s = 0.f;
    for (int d = 0; d < 256; d++) s += (float)q[d] * (float)ksh[j*256 + d];
    sc[i*64 + j] = s * 0.0625f; // 1/sqrt(256)
  }
  __syncthreads();
  if (tid < FBINS){
    int i = tid; float mx = -1e30f;
    for (int j = 0; j < FBINS; j++) mx = fmaxf(mx, sc[i*64 + j]);
    float sm = 0.f;
    for (int j = 0; j < FBINS; j++){ float e = __expf(sc[i*64 + j] - mx); sc[i*64 + j] = e; sm += e; }
    float inv = 1.f / sm;
    for (int j = 0; j < FBINS; j++) sc[i*64 + j] *= inv;
  }
  __syncthreads();
  if (tid < FBINS){
    int j = tid; float s = 0.f;
    for (int i = 0; i < FBINS; i++) s += sc[i*64 + j];
    wsum[j] = s;
  }
  __syncthreads();
  {
    int d = tid; // 0..255
    float s = 0.f;
    for (int j = 0; j < FBINS; j++) s += wsum[j] * (float)base[(size_t)j*S3 + 2*HDIM + h*256 + d];
    obar[(size_t)gb*HDIM + h*256 + d] = (__bf16)(s * (1.f / (float)FBINS));
  }
}

// ---------------- GRU cell pointwise ----------------
__global__ void k_gru_cell(const float* __restrict__ gi, const float* __restrict__ gh,
                           float* __restrict__ hf, __bf16* __restrict__ hb){
  int idx = blockIdx.x * 256 + threadIdx.x;
  int b = idx >> 10, i = idx & 1023;
  const float* gib = gi + (size_t)b * S3;
  const float* ghb = gh + (size_t)b * S3;
  float r = 1.f / (1.f + __expf(-(gib[i]          + ghb[i])));
  float z = 1.f / (1.f + __expf(-(gib[HDIM + i]   + ghb[HDIM + i])));
  float n = tanhf(gib[2*HDIM + i] + r * ghb[2*HDIM + i]);
  float hv = (1.f - z) * n + z * hf[idx];
  hf[idx] = hv;
  hb[idx] = (__bf16)hv;
}

// ---------------- gate + context mix, writes bf16 spl input base ----------------
__global__ void k_gate(const float* __restrict__ hidden, const float* __restrict__ gW,
                       const float* __restrict__ gb, const float* __restrict__ ctx,
                       __bf16* __restrict__ inp){
  int b = blockIdx.x, tid = threadIdx.x; // 256
  __shared__ float red[256]; __shared__ float gsh;
  float s = 0.f;
  for (int i = tid; i < HDIM; i += 256) s += hidden[(size_t)b*HDIM + i] * gW[i];
  red[tid] = s; __syncthreads();
  for (int off = 128; off > 0; off >>= 1){ if (tid < off) red[tid] += red[tid + off]; __syncthreads(); }
  if (tid == 0) gsh = tanhf(red[0] + gb[0]);
  __syncthreads();
  float g = gsh;
  for (int i = tid; i < HDIM; i += 256)
    inp[(size_t)b*INPK + i] = (__bf16)(hidden[(size_t)b*HDIM + i] + g * ctx[(size_t)b*HDIM + i]);
}

// ---------------- per-joint ancestor fill (cols 1024..1087) ----------------
__global__ void k_fill(__bf16* __restrict__ inp, const float* __restrict__ pred,
                       int na, int a0, int a1, int a2, int a3, int a4){
  int b = blockIdx.x, t = threadIdx.x; // 64
  float v = 0.f;
  if (t < na * 9){
    int ai = t / 9;
    int a = (ai == 0) ? a0 : (ai == 1) ? a1 : (ai == 2) ? a2 : (ai == 3) ? a3 : a4;
    v = pred[(size_t)b*DIM + a*9 + (t % 9)];
  }
  inp[(size_t)b*INPK + HDIM + t] = (__bf16)v;
}

// ---------------- x += delta, emit output, refresh bf16 x ----------------
__global__ void k_xstep(float* __restrict__ xf, __bf16* __restrict__ xb,
                        const float* __restrict__ pred, float* __restrict__ out, int t){
  int b = blockIdx.x, d = threadIdx.x; // 160
  if (d < DIM){
    float nx = xf[(size_t)b*DIM + d] + pred[(size_t)b*DIM + d];
    xf[(size_t)b*DIM + d] = nx;
    out[((size_t)b*PREDL + t)*DIM + d] = nx;
    xb[(size_t)b*DP + d] = (__bf16)nx;
  } else xb[(size_t)b*DP + d] = (__bf16)0.f;
}

// ---------------- host ----------------
static void gemm(hipStream_t st, const __bf16* A, int lda, const __bf16* W, int ldw,
                 const float* bias, float* Cf, int ldc, __bf16* Cb, int ldcb,
                 int M, int N, int K, int nvalid, int act){
  if (N >= 256) {                 // waves 2x4, block tile 64x256
    dim3 g(N / 256, M / 64);
    k_gemm<4><<<g, 256, 0, st>>>(A, lda, W, ldw, bias, Cf, ldc, Cb, ldcb, K, nvalid, act);
  } else {                        // N == 64: waves 8x1, block tile 256x64
    dim3 g(N / 64, M / 256);
    k_gemm<1><<<g, 256, 0, st>>>(A, lda, W, ldw, bias, Cf, ldc, Cb, ldcb, K, nvalid, act);
  }
}

static const int CH[15][5] = {
  {0,0,0,0,0},{0,0,0,0,0},{0,0,0,0,0},
  {0,0,0,0,0},{1,0,0,0,0},{2,0,0,0,0},
  {2,5,0,0,0},{2,5,6,0,0},{2,5,6,0,0},{2,5,6,0,0},
  {2,5,6,7,0},{2,5,6,8,0},{2,5,6,9,0},
  {2,5,6,7,10},{2,5,6,8,11}};
static const int CHN[15] = {0,0,0,1,1,1,2,3,3,3,4,4,4,5,5};

extern "C" void kernel_launch(void* const* d_in, const int* in_sizes, int n_in,
                              void* d_out, int out_size, void* d_ws, size_t ws_size,
                              hipStream_t stream) {
  (void)in_sizes; (void)n_in; (void)out_size; (void)ws_size;
  const float* poses   = (const float*)d_in[0];
  const float* freq_W  = (const float*)d_in[1];
  const float* freq_bv = (const float*)d_in[2];
  const float* attn_Wi = (const float*)d_in[3];
  const float* attn_bi = (const float*)d_in[4];
  const float* attn_Wo = (const float*)d_in[5];
  const float* attn_bo = (const float*)d_in[6];
  const float* gWih0   = (const float*)d_in[7];
  const float* gWhh0   = (const float*)d_in[8];
  const float* bih0    = (const float*)d_in[9];
  const float* bhh0    = (const float*)d_in[10];
  const float* gWih1   = (const float*)d_in[11];
  const float* gWhh1   = (const float*)d_in[12];
  const float* bih1    = (const float*)d_in[13];
  const float* bhh1    = (const float*)d_in[14];
  const float* pre_W   = (const float*)d_in[15];
  const float* pre_b   = (const float*)d_in[16];
  const float* gate_W  = (const float*)d_in[17];
  const float* gate_b  = (const float*)d_in[18];
  const float* spl_W1  = (const float*)d_in[19];
  const float* spl_b1  = (const float*)d_in[20];
  const float* spl_W2  = (const float*)d_in[21];
  const float* spl_b2  = (const float*)d_in[22];

  char* ws = (char*)d_ws;
  size_t off = 0;
  auto carve = [&](size_t bytes) -> char* {
    char* p = ws + off;
    off = (off + bytes + 255) & ~(size_t)255;
    return p;
  };
  __bf16* wFreq = (__bf16*)carve((size_t)HDIM*DP*2);
  __bf16* wAi   = (__bf16*)carve((size_t)S3*HDIM*2);
  __bf16* wAo   = (__bf16*)carve((size_t)HDIM*HDIM*2);
  __bf16* wIh0  = (__bf16*)carve((size_t)S3*DP*2);
  __bf16* wHh0  = (__bf16*)carve((size_t)S3*HDIM*2);
  __bf16* wIh1  = (__bf16*)carve((size_t)S3*HDIM*2);
  __bf16* wHh1  = (__bf16*)carve((size_t)S3*HDIM*2);
  __bf16* wPre  = (__bf16*)carve((size_t)HDIM*HDIM*2);
  __bf16* wS1   = (__bf16*)carve((size_t)15*64*INPK*2);
  __bf16* wS2   = (__bf16*)carve((size_t)15*64*64*2);
  __bf16* seedb = (__bf16*)carve((size_t)BATCH*SEEDL*DP*2);
  __bf16* freqb = (__bf16*)carve((size_t)BATCH*FBINS*DP*2);
  __bf16* featb = (__bf16*)carve((size_t)CHB*FBINS*HDIM*2);
  __bf16* qkvb  = (__bf16*)carve((size_t)CHB*FBINS*S3*2);
  __bf16* obarb = (__bf16*)carve((size_t)BATCH*HDIM*2);
  float*  ctx   = (float*) carve((size_t)BATCH*HDIM*4);
  float*  gi0   = (float*) carve((size_t)BATCH*S3*4);
  float*  gh0   = (float*) carve((size_t)BATCH*S3*4);
  float*  gi1   = (float*) carve((size_t)BATCH*S3*4);
  float*  gh1   = (float*) carve((size_t)BATCH*S3*4);
  float*  h0f   = (float*) carve((size_t)BATCH*HDIM*4);
  float*  h1f   = (float*) carve((size_t)BATCH*HDIM*4);
  __bf16* h0b   = (__bf16*)carve((size_t)BATCH*HDIM*2);
  __bf16* h1b   = (__bf16*)carve((size_t)BATCH*HDIM*2);
  float*  hidden= (float*) carve((size_t)BATCH*HDIM*4);
  __bf16* inp   = (__bf16*)carve((size_t)BATCH*INPK*2);
  __bf16* hidb  = (__bf16*)carve((size_t)BATCH*64*2);
  float*  pred  = (float*) carve((size_t)BATCH*DIM*4);
  float*  xf    = (float*) carve((size_t)BATCH*DIM*4);
  __bf16* xb    = (__bf16*)carve((size_t)BATCH*DP*2);
  float* outp = (float*)d_out;

  // ---- prep: weights -> bf16 (K-padded, N-major) ----
  k_cvt_pad<<<HDIM, 256, 0, stream>>>(wFreq, freq_W, DIM, DP);
  k_cvt_pad<<<S3,   256, 0, stream>>>(wAi,   attn_Wi, HDIM, HDIM);
  k_cvt_pad<<<HDIM, 256, 0, stream>>>(wAo,   attn_Wo, HDIM, HDIM);
  k_cvt_pad<<<S3,   256, 0, stream>>>(wIh0,  gWih0, DIM, DP);
  k_cvt_pad<<<S3,   256, 0, stream>>>(wHh0,  gWhh0, HDIM, HDIM);
  k_cvt_pad<<<S3,   256, 0, stream>>>(wIh1,  gWih1, HDIM, HDIM);
  k_cvt_pad<<<S3,   256, 0, stream>>>(wHh1,  gWhh1, HDIM, HDIM);
  k_cvt_pad<<<HDIM, 256, 0, stream>>>(wPre,  pre_W, HDIM, HDIM);
  k_cvt_w1<<<15*64, 256, 0, stream>>>(wS1, spl_W1);
  k_cvt_w2<<<15*64, 64,  0, stream>>>(wS2, spl_W2);
  k_cvt_seed<<<BATCH*SEEDL, DP, 0, stream>>>(seedb, poses);
  k_zero_f32<<<(BATCH*HDIM+255)/256, 256, 0, stream>>>(h0f, BATCH*HDIM);
  k_zero_f32<<<(BATCH*HDIM+255)/256, 256, 0, stream>>>(h1f, BATCH*HDIM);
  k_zero_b16<<<(BATCH*HDIM+255)/256, 256, 0, stream>>>(h0b, BATCH*HDIM);
  k_zero_b16<<<(BATCH*HDIM+255)/256, 256, 0, stream>>>(h1b, BATCH*HDIM);
  k_init_x<<<BATCH, DP, 0, stream>>>(xf, xb, poses);

  // ---- rfft.real + attention (chunked over batch), pooled context ----
  k_freq<<<BATCH*FBINS, DP, 0, stream>>>(poses, freqb);
  for (int c = 0; c < BATCH/CHB; c++){
    const __bf16* fq = freqb + (size_t)c*CHB*FBINS*DP;
    int Mc = CHB * FBINS; // 3904
    gemm(stream, fq,    DP,   wFreq, DP,   freq_bv, nullptr, 0, featb, HDIM, Mc, HDIM, DP,   HDIM, 0);
    gemm(stream, featb, HDIM, wAi,   HDIM, attn_bi, nullptr, 0, qkvb,  S3,   Mc, S3,   HDIM, S3,   0);
    k_attn<<<CHB*4, 256, 0, stream>>>(qkvb, obarb, c*CHB);
  }
  gemm(stream, obarb, HDIM, wAo, HDIM, attn_bo, ctx, HDIM, nullptr, 0, BATCH, HDIM, HDIM, HDIM, 0);

  // ---- encoder: 120 GRU steps ----
  for (int t = 0; t < SEEDL; t++){
    gemm(stream, seedb + (size_t)t*DP, SEEDL*DP, wIh0, DP,   bih0, gi0, S3, nullptr, 0, BATCH, S3, DP,   S3, 0);
    gemm(stream, h0b,   HDIM,          wHh0,     HDIM, bhh0, gh0, S3, nullptr, 0, BATCH, S3, HDIM, S3, 0);
    k_gru_cell<<<BATCH*HDIM/256, 256, 0, stream>>>(gi0, gh0, h0f, h0b);
    gemm(stream, h0b, HDIM, wIh1, HDIM, bih1, gi1, S3, nullptr, 0, BATCH, S3, HDIM, S3, 0);
    gemm(stream, h1b, HDIM, wHh1, HDIM, bhh1, gh1, S3, nullptr, 0, BATCH, S3, HDIM, S3, 0);
    k_gru_cell<<<BATCH*HDIM/256, 256, 0, stream>>>(gi1, gh1, h1f, h1b);
  }

  // ---- decoder: 24 autoregressive steps ----
  for (int t = 0; t < PREDL; t++){
    gemm(stream, xb,  DP,   wIh0, DP,   bih0, gi0, S3, nullptr, 0, BATCH, S3, DP,   S3, 0);
    gemm(stream, h0b, HDIM, wHh0, HDIM, bhh0, gh0, S3, nullptr, 0, BATCH, S3, HDIM, S3, 0);
    k_gru_cell<<<BATCH*HDIM/256, 256, 0, stream>>>(gi0, gh0, h0f, h0b);
    gemm(stream, h0b, HDIM, wIh1, HDIM, bih1, gi1, S3, nullptr, 0, BATCH, S3, HDIM, S3, 0);
    gemm(stream, h1b, HDIM, wHh1, HDIM, bhh1, gh1, S3, nullptr, 0, BATCH, S3, HDIM, S3, 0);
    k_gru_cell<<<BATCH*HDIM/256, 256, 0, stream>>>(gi1, gh1, h1f, h1b);

    gemm(stream, h1b, HDIM, wPre, HDIM, pre_b, hidden, HDIM, nullptr, 0, BATCH, HDIM, HDIM, HDIM, 1);
    k_gate<<<BATCH, 256, 0, stream>>>(hidden, gate_W, gate_b, ctx, inp);

    for (int j = 0; j < 15; j++){
      k_fill<<<BATCH, 64, 0, stream>>>(inp, pred, CHN[j], CH[j][0], CH[j][1], CH[j][2], CH[j][3], CH[j][4]);
      gemm(stream, inp,  INPK, wS1 + (size_t)j*64*INPK, INPK, spl_b1 + j*64, nullptr, 0, hidb, 64,
           BATCH, 64, INPK, 64, 1);
      gemm(stream, hidb, 64,   wS2 + (size_t)j*64*64,   64,   spl_b2 + j*9,  pred + j*9, DIM, nullptr, 0,
           BATCH, 64, 64, 9, 2);
    }
    k_xstep<<<BATCH, DP, 0, stream>>>(xf, xb, pred, outp, t);
  }
}
